// SingleHeadAttentionModule_82411832476079
// MI455X (gfx1250) — compile-verified, matrix operand order checked by reading, not run on hardware
//
#include <hip/hip_runtime.h>

// ---------------------------------------------------------------------------
// Single-head causal attention for MI455X (gfx1250, wave32, WMMA bf16).
//   q = (query @ wq) * 1/sqrt(H)   (scale folded into wq)
//   k =  key   @ wk
//   v =  value @ wv   (stored transposed [b][h][s] for contiguous PV B-frags)
//   out = softmax(causal(q k^T)) @ v      computed flash-attention style.
// All matmuls use v_wmma_f32_16x16x32_bf16. q/k/v (6 MB bf16) live in L2.
// ---------------------------------------------------------------------------

typedef __bf16 bf16_t;
typedef __attribute__((ext_vector_type(16))) __bf16 v16bf;
typedef __attribute__((ext_vector_type(8)))  __bf16 v8bf;
typedef __attribute__((ext_vector_type(8)))  float  v8f;
typedef __attribute__((ext_vector_type(4)))  float  v4f;

#define EMBED 1024
#define HEAD  64
#define BATCH 4
#define SEQ   4096
#define BS    (BATCH * SEQ)

// ws layout in bf16 elements:
#define WFRAG_ELEMS 65536                 // 32 kblk * 4 ntile * 32 lanes * 16
#define QOFF (3 * WFRAG_ELEMS)            // q  [BS][64]
#define KOFF (QOFF + BS * HEAD)           // k  [BS][64]
#define VOFF (KOFF + BS * HEAD)           // v^T [B][64][4096]
// total = 196608 + 3*1048576 = 3342336 elems = ~6.4 MB

__device__ inline v16bf mk16(v8bf lo, v8bf hi) {
    v16bf r;
#pragma unroll
    for (int i = 0; i < 8; ++i) { r[i] = lo[i]; r[i + 8] = hi[i]; }
    return r;
}

// ---------------------------------------------------------------------------
// Kernel 1: fp32 weights -> bf16, pre-swizzled into WMMA B-fragment order.
// B-frag (32x16, bf16): lane L holds col N=L&15; K = (L>>4)*8 + {0..7,16..23}.
// ---------------------------------------------------------------------------
__global__ void prep_weights(const float* __restrict__ wq,
                             const float* __restrict__ wk,
                             const float* __restrict__ wv,
                             bf16_t* __restrict__ ws) {
    int idx = blockIdx.x * 256 + threadIdx.x;
    if (idx >= 3 * WFRAG_ELEMS) return;
    int i    =  idx        & 15;
    int lane = (idx >> 4)  & 31;
    int nt   = (idx >> 9)  & 3;
    int kblk = (idx >> 11) & 31;
    int w    =  idx >> 16;
    const float* W = (w == 0) ? wq : ((w == 1) ? wk : wv);
    int col = nt * 16 + (lane & 15);
    int kb  = kblk * 32 + ((lane >> 4) * 8);
    int K   = (i < 8) ? (kb + i) : (kb + 16 + (i - 8));
    float val = W[K * HEAD + col];
    if (w == 0) val *= 0.125f;            // fold 1/sqrt(64) into wq
    ws[w * WFRAG_ELEMS + (idx & (WFRAG_ELEMS - 1))] = (bf16_t)val;
}

// ---------------------------------------------------------------------------
// Kernel 2: projections. One wave per 16-row tile; grid.y selects q/k/v.
// ---------------------------------------------------------------------------
__global__ void __launch_bounds__(32)
proj(const float* __restrict__ query, const float* __restrict__ key,
     const float* __restrict__ value, bf16_t* __restrict__ ws) {
    const int tile = blockIdx.x;          // 0 .. BS/16-1
    const int mat  = blockIdx.y;          // 0=q 1=k 2=v
    const int lane = threadIdx.x;
    const float*  X  = (mat == 0) ? query : ((mat == 1) ? key : value);
    const bf16_t* wf = ws + mat * WFRAG_ELEMS;

    const int row  = lane & 15;           // A-layout: M = lane&15
    const int kb8  = (lane >> 4) * 8;     // K sub-offset per half-wave
    const float* xrow = X + (size_t)(tile * 16 + row) * EMBED + kb8;

    v8f acc[4] = {};                      // 4 N-tiles of 16 (H = 64)

    for (int k0 = 0; k0 < EMBED; k0 += 32) {
        const float* p = xrow + k0;
        v4f f0 = *(const v4f*)(p);
        v4f f1 = *(const v4f*)(p + 4);
        v4f f2 = *(const v4f*)(p + 16);
        v4f f3 = *(const v4f*)(p + 20);
        v16bf a;
#pragma unroll
        for (int i = 0; i < 4; ++i) {
            a[i]      = (bf16_t)f0[i];
            a[i + 4]  = (bf16_t)f1[i];
            a[i + 8]  = (bf16_t)f2[i];
            a[i + 12] = (bf16_t)f3[i];
        }
        const int kblk = k0 >> 5;
#pragma unroll
        for (int nt = 0; nt < 4; ++nt) {
            const v8bf* bp = (const v8bf*)(wf + ((kblk * 4 + nt) * 32 + lane) * 16);
            v16bf bfrag = mk16(bp[0], bp[1]);
            acc[nt] = __builtin_amdgcn_wmma_f32_16x16x32_bf16(
                false, a, false, bfrag, (short)0, acc[nt], false, false);
        }
    }

    // store C-layout (lane: N = lane&15; VGPR r -> M = r + 8*(lane>=16))
    const int half = lane >> 4;
    const int nloc = lane & 15;
#pragma unroll
    for (int nt = 0; nt < 4; ++nt) {
#pragma unroll
        for (int r = 0; r < 8; ++r) {
            int m    = r + half * 8;
            int grow = tile * 16 + m;
            int col  = nt * 16 + nloc;
            float val = acc[nt][r];
            if (mat == 2) {               // v stored transposed [b][h][s]
                int b = grow >> 12, s = grow & 4095;
                ws[VOFF + ((b * HEAD + col) << 12) + s] = (bf16_t)val;
            } else {
                bf16_t* out = ws + (mat == 0 ? QOFF : KOFF);
                out[grow * HEAD + col] = (bf16_t)val;
            }
        }
    }
}

// ---------------------------------------------------------------------------
// Kernel 3: flash attention. One wave per (batch, 16-query tile).
// ---------------------------------------------------------------------------
__global__ void __launch_bounds__(32)
attn(const bf16_t* __restrict__ ws, float* __restrict__ out) {
    const int qt   = blockIdx.x;          // 0 .. 255
    const int b    = blockIdx.y;          // 0 .. 3
    const int lane = threadIdx.x;
    const bf16_t* qbf = ws + QOFF;
    const bf16_t* kbf = ws + KOFF;
    const bf16_t* vt  = ws + VOFF + (size_t)b * HEAD * SEQ;

    __shared__ bf16_t pbuf[16 * 32];      // P tile, row-major [m][kk]

    const int nloc = lane & 15;
    const int half = lane >> 4;
    const int kb8  = half * 8;

    // q A-fragments (16x32 each) for h 0..31 and 32..63; row M = lane&15
    v16bf qa[2];
    {
        const bf16_t* qrow = qbf + (size_t)(b * SEQ + qt * 16 + nloc) * HEAD;
#pragma unroll
        for (int hb = 0; hb < 2; ++hb) {
            const bf16_t* p0 = qrow + hb * 32 + kb8;
            qa[hb] = mk16(*(const v8bf*)p0, *(const v8bf*)(p0 + 16));
        }
    }

    v8f acc[4] = {};                      // O tiles, 4 x (16x16) f32
    float mi[8], li[8];
#pragma unroll
    for (int r = 0; r < 8; ++r) { mi[r] = -1e30f; li[r] = 0.0f; }

    const int nblk = (qt + 2) >> 1;       // 32-key blocks covering causal extent
    for (int blk = 0; blk < nblk; ++blk) {
        const int key0 = blk * 32;

        // scores: S[t] = q (16x64) . k^T (64x16), two 16-key sub-tiles
        v8f s[2] = {};
#pragma unroll
        for (int t = 0; t < 2; ++t) {
            const bf16_t* krow =
                kbf + (size_t)(b * SEQ + key0 + t * 16 + nloc) * HEAD + kb8;
#pragma unroll
            for (int hb = 0; hb < 2; ++hb) {
                v16bf kfrag = mk16(*(const v8bf*)(krow + hb * 32),
                                   *(const v8bf*)(krow + hb * 32 + 16));
                s[t] = __builtin_amdgcn_wmma_f32_16x16x32_bf16(
                    false, qa[hb], false, kfrag, (short)0, s[t], false, false);
            }
        }

        // online softmax per row (rows split: lanes 0-15 -> M=r, 16-31 -> M=r+8)
#pragma unroll
        for (int r = 0; r < 8; ++r) {
            const int m  = r + half * 8;
            const int qg = qt * 16 + m;
            float v0 = (key0 + nloc      <= qg) ? s[0][r] : -1e30f;
            float v1 = (key0 + 16 + nloc <= qg) ? s[1][r] : -1e30f;
            float rmax = fmaxf(v0, v1);
#pragma unroll
            for (int d = 1; d < 16; d <<= 1)
                rmax = fmaxf(rmax, __shfl_xor(rmax, d, 32));
            float mnew  = fmaxf(mi[r], rmax);
            float alpha = __expf(mi[r] - mnew);
            float p0 = __expf(v0 - mnew);
            float p1 = __expf(v1 - mnew);
            float rsum = p0 + p1;
#pragma unroll
            for (int d = 1; d < 16; d <<= 1)
                rsum += __shfl_xor(rsum, d, 32);
            li[r] = li[r] * alpha + rsum;
            mi[r] = mnew;
#pragma unroll
            for (int ht = 0; ht < 4; ++ht) acc[ht][r] *= alpha;
            pbuf[m * 32 + nloc]      = (bf16_t)p0;   // C-layout -> row-major LDS
            pbuf[m * 32 + 16 + nloc] = (bf16_t)p1;
        }
        __syncthreads();                  // order ds_store -> ds_load

        // reload P as an A-fragment (16x32) and do O += P @ V
        const bf16_t* pr = pbuf + nloc * 32 + kb8;   // row M = lane&15
        v16bf pa = mk16(*(const v8bf*)pr, *(const v8bf*)(pr + 16));
#pragma unroll
        for (int ht = 0; ht < 4; ++ht) {
            const bf16_t* vrow = vt + (size_t)(ht * 16 + nloc) * SEQ + key0 + kb8;
            v16bf vfrag = mk16(*(const v8bf*)vrow, *(const v8bf*)(vrow + 16));
            acc[ht] = __builtin_amdgcn_wmma_f32_16x16x32_bf16(
                false, pa, false, vfrag, (short)0, acc[ht], false, false);
        }
        __syncthreads();                  // WAR guard before next P store
    }

    // epilogue: out = acc / l
#pragma unroll
    for (int r = 0; r < 8; ++r) {
        const int m   = r + half * 8;
        const int row = b * SEQ + qt * 16 + m;
        const float inv = 1.0f / li[r];
#pragma unroll
        for (int ht = 0; ht < 4; ++ht)
            out[(size_t)row * HEAD + ht * 16 + nloc] = acc[ht][r] * inv;
    }
}

// ---------------------------------------------------------------------------
extern "C" void kernel_launch(void* const* d_in, const int* in_sizes, int n_in,
                              void* d_out, int out_size, void* d_ws, size_t ws_size,
                              hipStream_t stream) {
    const float* query = (const float*)d_in[0];
    const float* key   = (const float*)d_in[1];
    const float* value = (const float*)d_in[2];
    const float* wq    = (const float*)d_in[3];
    const float* wk    = (const float*)d_in[4];
    const float* wv    = (const float*)d_in[5];
    bf16_t* ws  = (bf16_t*)d_ws;          // needs ~6.4 MB
    float*  out = (float*)d_out;

    prep_weights<<<(3 * WFRAG_ELEMS + 255) / 256, 256, 0, stream>>>(wq, wk, wv, ws);
    proj<<<dim3(BS / 16, 3), 32, 0, stream>>>(query, key, value, ws);
    attn<<<dim3(SEQ / 16, BATCH), 32, 0, stream>>>(ws, out);
}
